// GINE_41635412967957
// MI455X (gfx1250) — matrix operand order repaired and checked by reading, not compiled, not run on hardware
//
#include <hip/hip_runtime.h>
#include <hip/hip_bf16.h>
#include <math.h>

// ---------------------------------------------------------------------------
// GINE forward for MI455X (gfx1250, wave32).
// v_wmma_f32_16x16x32_bf16 for all GEMMs (fp32 accumulate), fp32 L2 atomics
// for the scatter-aggregate. 32-bit byte-offset addressing (saddr+voffset),
// b128 edge-index loads, native f32->bf16 converts, transposed-LDS weights
// for vector ds loads, batched gathers, global_prefetch for edge_attr.
// ---------------------------------------------------------------------------

typedef __attribute__((ext_vector_type(16))) __bf16 v16bf;
typedef __attribute__((ext_vector_type(8)))  __bf16 v8bf;
typedef __attribute__((ext_vector_type(8)))  float  v8f;
typedef __attribute__((ext_vector_type(4)))  int    v4i;

#define N_NODES  50000
#define N_EDGES  800000
#define F_IN     32
#define DHID     64
#define EDGE_DIM 16
#define NLAYERS  3
#define DENSE_W  256
#define N_GRAPHS 256

// Native conversion: lets the backend use the hardware f32->bf16 cvt ops
// instead of a 3-op integer round-to-nearest-even sequence.
__device__ __forceinline__ __bf16 f2bf(float f) { return (__bf16)f; }

__device__ __forceinline__ v8f wmma_bf16(v16bf a, v16bf b, v8f c) {
    // (neg_a, A, neg_b, B, c_mod, C, reuse_a, reuse_b)
    return __builtin_amdgcn_wmma_f32_16x16x32_bf16(false, a, false, b,
                                                   (short)0, c, false, false);
}

// 32-bit byte-offset loads/atomics: base stays uniform (SGPR pair), offset is
// a zero-extended 32-bit VGPR -> saddr+voffset global addressing.
__device__ __forceinline__ float ld_off(const float* base, unsigned boff) {
    return *(const float*)((const char*)base + boff);
}
__device__ __forceinline__ float* ptr_off(float* base, unsigned boff) {
    return (float*)((char*)base + boff);
}

// ---------------------------------------------------------------------------
__global__ void fill_zero_kernel(float* __restrict__ p, int n) {
    int i = blockIdx.x * blockDim.x + threadIdx.x;
    int stride = gridDim.x * blockDim.x;
    for (; i < n; i += stride) p[i] = 0.0f;
}

// ---------------------------------------------------------------------------
// Fused edge message + aggregate:
//   m = relu(x[src] + edge_attr @ We + be); agg[dst] += m
// One wave per 16-edge tile; A = edge_attr (16x16, K zero-padded to 32),
// B = We staged transposed in LDS. C fragments scattered with fp32 atomics.
// ---------------------------------------------------------------------------
template<int EOUT>
__global__ __launch_bounds__(256) void edge_msg_kernel(
    const float* __restrict__ xin,     // N_NODES x EOUT
    const float* __restrict__ eattr,   // N_EDGES x EDGE_DIM
    const int*   __restrict__ srcIdx,
    const int*   __restrict__ dstIdx,
    const float* __restrict__ We,      // EDGE_DIM x EOUT (row-major K x N)
    const float* __restrict__ be,      // EOUT
    float*       __restrict__ agg,     // N_NODES x EOUT
    int nTiles)
{
    constexpr int NT = EOUT / 16;
    // Transposed: sWeT[n * EDGE_DIM + k]  (each n-row = 16 bf16 = 32B, aligned)
    __shared__ __bf16 sWeT[EOUT * EDGE_DIM];
    for (int i = threadIdx.x; i < EOUT * EDGE_DIM; i += blockDim.x) {
        int n = i >> 4, k = i & 15;
        sWeT[i] = f2bf(We[k * EOUT + n]);
    }
    __syncthreads();

    const int lane   = threadIdx.x & 31;
    const int col    = lane & 15;
    const int hiHalf = (lane >> 4) & 1;
    const int kb     = hiHalf * 8;    // A fragment K offset
    const int mOff   = hiHalf * 8;    // C fragment row offset

    v16bf bz;
    #pragma unroll
    for (int i = 0; i < 16; ++i) bz[i] = f2bf(0.0f);

    // B fragments (loop invariant). K padded 16->32: hi half-wave (K=16..31)
    // is all zeros; lo half-wave reads one contiguous 32B column of We^T.
    v16bf bfrag[NT];
    #pragma unroll
    for (int t = 0; t < NT; ++t) {
        v16bf bv = *(const v16bf*)&sWeT[(t * 16 + col) * EDGE_DIM];
        if (hiHalf) bv = bz;
        bfrag[t] = bv;
    }
    float bias[NT];
    #pragma unroll
    for (int t = 0; t < NT; ++t) bias[t] = be[t * 16 + col];

    int gw      = blockIdx.x * (blockDim.x >> 5) + (threadIdx.x >> 5);
    int gstride = gridDim.x * (blockDim.x >> 5);

    for (int tile = gw; tile < nTiles; tile += gstride) {
        const int ebase = tile * 16;

        // prefetch next tile's edge_attr stream (global_prefetch_b8)
        if (tile + gstride < nTiles) {
            __builtin_prefetch(eattr +
                (size_t)((tile + gstride) * 16 + col) * EDGE_DIM, 0, 0);
        }

        // A fragment: row = col, contiguous K 0..15 attrs (hi half-wave multiplies zero B).
        v16bf a;
        const float* ea = eattr + (size_t)(ebase + col) * EDGE_DIM;
        #pragma unroll
        for (int i = 0; i < 8; ++i) a[i] = f2bf(ea[i]);
        #pragma unroll
        for (int i = 0; i < 8; ++i) a[8 + i] = f2bf(ea[8 + i]);

        // This lane's 8 src/dst indices: contiguous, 16B aligned -> 2x b128 each
        const v4i* sp = (const v4i*)(srcIdx + ebase + mOff);
        const v4i* dp = (const v4i*)(dstIdx + ebase + mOff);
        v4i s01 = sp[0], s23 = sp[1];
        v4i d01 = dp[0], d23 = dp[1];
        unsigned srB[8], drB[8];   // row base byte-offsets into x / agg
        #pragma unroll
        for (int r = 0; r < 4; ++r) {
            srB[r]     = (unsigned)s01[r] * (unsigned)(EOUT * 4);
            srB[r + 4] = (unsigned)s23[r] * (unsigned)(EOUT * 4);
            drB[r]     = (unsigned)d01[r] * (unsigned)(EOUT * 4);
            drB[r + 4] = (unsigned)d23[r] * (unsigned)(EOUT * 4);
        }

        // all WMMAs first
        v8f cfr[NT];
        #pragma unroll
        for (int t = 0; t < NT; ++t) {
            v8f c = {};
            cfr[t] = wmma_bf16(a, bfrag[t], c);
        }

        // batched gathers of x[src] (issue all loads, single wait)
        float xv[NT][8];
        #pragma unroll
        for (int t = 0; t < NT; ++t) {
            const unsigned fB = (unsigned)((t * 16 + col) * 4);
            #pragma unroll
            for (int r = 0; r < 8; ++r)
                xv[t][r] = ld_off(xin, srB[r] + fB);
        }

        // relu + scatter atomics
        #pragma unroll
        for (int t = 0; t < NT; ++t) {
            const unsigned fB = (unsigned)((t * 16 + col) * 4);
            #pragma unroll
            for (int r = 0; r < 8; ++r) {
                float v = cfr[t][r] + bias[t] + xv[t][r];
                v = fmaxf(v, 0.0f);
                atomicAdd(ptr_off(agg, drB[r] + fB), v);
            }
        }
    }
}

// ---------------------------------------------------------------------------
// Fused node MLP: hout = relu( relu(((1+eps)*h + agg) @ W1 + b1) @ W2 + b2 )
// One wave per 16-node tile; weights staged transposed in LDS (vector ds
// loads), intermediate staged in per-wave LDS (s_wait_dscnt 0 for RAW).
// ---------------------------------------------------------------------------
template<int DIN>
__global__ __launch_bounds__(256) void node_mlp_kernel(
    const float* __restrict__ hin,   // N x DIN
    const float* __restrict__ agg,   // N x DIN
    const float* __restrict__ epsp,  // scalar
    const float* __restrict__ W1,    // DIN x 64
    const float* __restrict__ b1,    // 64
    const float* __restrict__ W2,    // 64 x 64
    const float* __restrict__ b2,    // 64
    float*       __restrict__ hout,  // N x 64
    int nTiles)
{
    constexpr int KT1 = DIN / 32;
    __shared__ __bf16 sW1T[64 * DIN];   // transposed: [n][k]
    __shared__ __bf16 sW2T[64 * 64];    // transposed: [n][k]
    __shared__ __bf16 sT[8][16 * 64];   // per-wave intermediate

    for (int i = threadIdx.x; i < 64 * DIN; i += blockDim.x) {
        int n = i / DIN, k = i % DIN;
        sW1T[i] = f2bf(W1[k * 64 + n]);
    }
    for (int i = threadIdx.x; i < 64 * 64; i += blockDim.x) {
        int n = i >> 6, k = i & 63;
        sW2T[i] = f2bf(W2[k * 64 + n]);
    }
    __syncthreads();

    const int warp   = threadIdx.x >> 5;
    const int lane   = threadIdx.x & 31;
    const int col    = lane & 15;
    const int hiHalf = (lane >> 4) & 1;
    const int koff   = hiHalf * 16;
    const int kb     = hiHalf * 8;
    const int mOff   = hiHalf * 8;

    const int gw = blockIdx.x * 8 + warp;     // wave-uniform
    if (gw >= nTiles) return;                 // whole-wave exit; EXEC stays full

    const float eps1  = 1.0f + epsp[0];
    const int rowBase = gw * 16;
    const unsigned rowB = (unsigned)(rowBase + col) * (unsigned)(DIN * 4);

    // A fragments of hprime = (1+eps)*h + agg (contiguous 32B chunks -> b128s)
    v16bf aF[KT1];
    #pragma unroll
    for (int kt = 0; kt < KT1; ++kt) {
        #pragma unroll
        for (int i = 0; i < 8; ++i) {
            unsigned o0 = rowB + (unsigned)((kt * 32 + koff + i) * 4);
            unsigned o1 = rowB + (unsigned)((kt * 32 + koff + 8 + i) * 4);
            aF[kt][i]     = f2bf(eps1 * ld_off(hin, o0) + ld_off(agg, o0));
            aF[kt][8 + i] = f2bf(eps1 * ld_off(hin, o1) + ld_off(agg, o1));
        }
    }

    __bf16* tb = &sT[warp][0];

    // GEMM1: T = relu(A @ W1 + b1), staged to per-wave LDS
    #pragma unroll
    for (int t = 0; t < 4; ++t) {
        v8f c = {};
        #pragma unroll
        for (int kt = 0; kt < KT1; ++kt) {
            // column of W1^T: 16 contiguous bf16 (32B, aligned) -> ds_load_b128
            v16bf b = *(const v16bf*)
                &sW1T[(t * 16 + col) * DIN + kt * 32 + koff];
            c = wmma_bf16(aF[kt], b, c);
        }
        const float bias = b1[t * 16 + col];
        #pragma unroll
        for (int r = 0; r < 8; ++r) {
            int m = mOff + r;
            tb[m * 64 + t * 16 + col] = f2bf(fmaxf(c[r] + bias, 0.0f));
        }
    }
    // intra-wave LDS RAW: wait for our DS stores before cross-lane reads
    asm volatile("s_wait_dscnt 0" ::: "memory");

    // Re-load T as A fragments (two aligned 16B chunks per fragment)
    v16bf a2[2];
    #pragma unroll
    for (int kt = 0; kt < 2; ++kt) {
        union { v16bf v; struct { v8bf lo, hi; } p; } u;
        u.p.lo = *(const v8bf*)&tb[col * 64 + kt * 32 + koff];
        u.p.hi = *(const v8bf*)&tb[col * 64 + kt * 32 + koff + 8];
        a2[kt] = u.v;
    }

    // GEMM2: out = relu(T @ W2 + b2)  (outer relu of the reference fused in)
    #pragma unroll
    for (int t = 0; t < 4; ++t) {
        v8f c = {};
        #pragma unroll
        for (int kt = 0; kt < 2; ++kt) {
            v16bf b = *(const v16bf*)
                &sW2T[(t * 16 + col) * 64 + kt * 32 + koff];
            c = wmma_bf16(a2[kt], b, c);
        }
        const float bias = b2[t * 16 + col];
        #pragma unroll
        for (int r = 0; r < 8; ++r) {
            unsigned oB = (unsigned)(rowBase + mOff + r) * 256u
                        + (unsigned)((t * 16 + col) * 4);
            *ptr_off(hout, oB) = fmaxf(c[r] + bias, 0.0f);
        }
    }
}

// ---------------------------------------------------------------------------
// Per-graph max/sum pooling. h >= 0 (post-relu), so float max == uint-bits max
// and a 0 init matches where(counts>0, gmax, 0).
// ---------------------------------------------------------------------------
__global__ void pool_kernel(const float* __restrict__ h,
                            const int*   __restrict__ batch,
                            unsigned*    __restrict__ gmaxU,
                            float*       __restrict__ gsum,
                            float*       __restrict__ counts)
{
    int idx = blockIdx.x * blockDim.x + threadIdx.x;
    int stride = gridDim.x * blockDim.x;
    for (; idx < N_NODES * DHID; idx += stride) {
        int node = idx >> 6;
        int f    = idx & 63;
        int g    = batch[node];
        float v  = h[idx];
        atomicMax(&gmaxU[g * DHID + f], __float_as_uint(v));
        atomicAdd(&gsum[g * DHID + f], v);
        if (f == 0) atomicAdd(&counts[g], 1.0f);
    }
}

__global__ void pool_finalize_kernel(const unsigned* __restrict__ gmaxU,
                                     const float* __restrict__ gsum,
                                     const float* __restrict__ counts,
                                     float* __restrict__ g)
{
    int idx = blockIdx.x * blockDim.x + threadIdx.x;
    if (idx >= N_GRAPHS * DHID) return;
    int gi = idx >> 6, f = idx & 63;
    float cnt = counts[gi];
    g[gi * (2 * DHID) + f]        = __uint_as_float(gmaxU[idx]);
    g[gi * (2 * DHID) + DHID + f] = gsum[idx] / fmaxf(cnt, 1.0f);
}

// ---------------------------------------------------------------------------
// Dense head: out = sigmoid((g @ Wl + bl) @ Wo + bo). One block per graph.
// ---------------------------------------------------------------------------
__global__ __launch_bounds__(256) void head_kernel(
    const float* __restrict__ g,   // N_GRAPHS x 128
    const float* __restrict__ Wl,  // 128 x 256
    const float* __restrict__ bl,  // 256
    const float* __restrict__ Wo,  // 256 x 1
    const float* __restrict__ bo,  // 1
    float* __restrict__ out)
{
    __shared__ float red[256];
    const int gi = blockIdx.x;
    const int j  = threadIdx.x;
    const float* gr = g + gi * (2 * DHID);
    float s = bl[j];
    #pragma unroll 4
    for (int k = 0; k < 2 * DHID; ++k) s += gr[k] * Wl[k * DENSE_W + j];
    red[j] = s * Wo[j];
    __syncthreads();
    for (int off = 128; off > 0; off >>= 1) {
        if (j < off) red[j] += red[j + off];
        __syncthreads();
    }
    if (j == 0) out[gi] = 1.0f / (1.0f + expf(-(red[0] + bo[0])));
}

// ---------------------------------------------------------------------------
extern "C" void kernel_launch(void* const* d_in, const int* in_sizes, int n_in,
                              void* d_out, int out_size, void* d_ws, size_t ws_size,
                              hipStream_t stream)
{
    const float* x     = (const float*)d_in[0];
    const float* eattr = (const float*)d_in[1];
    const int*   eidx  = (const int*)  d_in[2];
    const int*   batch = (const int*)  d_in[3];
    const float* eps1  = (const float*)d_in[4];
    const float* We1   = (const float*)d_in[5];
    const float* be1   = (const float*)d_in[6];
    const float* W1a   = (const float*)d_in[7];
    const float* b1a   = (const float*)d_in[8];
    const float* W1b   = (const float*)d_in[9];
    const float* b1b   = (const float*)d_in[10];
    const float* epsL  = (const float*)d_in[11];
    const float* WeL   = (const float*)d_in[12];
    const float* beL   = (const float*)d_in[13];
    const float* WLa   = (const float*)d_in[14];
    const float* bLa   = (const float*)d_in[15];
    const float* WLb   = (const float*)d_in[16];
    const float* bLb   = (const float*)d_in[17];
    const float* Wl    = (const float*)d_in[18];
    const float* bl    = (const float*)d_in[19];
    const float* Wo    = (const float*)d_in[20];
    const float* bo    = (const float*)d_in[21];
    float* out = (float*)d_out;

    const int* srcI = eidx;             // edge_index[0]
    const int* dstI = eidx + N_EDGES;   // edge_index[1]

    // workspace layout (all fp32)
    char* ws = (char*)d_ws;
    float*    hA     = (float*)(ws);
    float*    hB     = (float*)(ws + (size_t)N_NODES * 64 * 4);
    float*    agg    = (float*)(ws + (size_t)N_NODES * 64 * 4 * 2);
    char*     tail   = ws + (size_t)N_NODES * 64 * 4 * 3;
    unsigned* gmaxU  = (unsigned*)tail;                                   // 256*64
    float*    gsum   = (float*)(tail + (size_t)N_GRAPHS * 64 * 4);        // 256*64
    float*    counts = (float*)(tail + (size_t)N_GRAPHS * 64 * 4 * 2);    // 256
    float*    gmat   = (float*)(tail + (size_t)N_GRAPHS * 64 * 4 * 2
                                     + (size_t)N_GRAPHS * 4);             // 256*128

    const int edgeTiles = N_EDGES / 16;   // 50000
    const int nodeTiles = N_NODES / 16;   // 3125
    const int mlpBlocks = (nodeTiles + 7) / 8;

    // ---- conv1 (F_IN = 32) ----
    fill_zero_kernel<<<256, 256, 0, stream>>>(agg, N_NODES * F_IN);
    edge_msg_kernel<32><<<1024, 256, 0, stream>>>(x, eattr, srcI, dstI,
                                                  We1, be1, agg, edgeTiles);
    node_mlp_kernel<32><<<mlpBlocks, 256, 0, stream>>>(x, agg, eps1,
                                                       W1a, b1a, W1b, b1b,
                                                       hA, nodeTiles);

    // ---- 3 GINEConv layers (64 -> 64) ----
    float* hc = hA; float* hn = hB;
    for (int i = 0; i < NLAYERS; ++i) {
        fill_zero_kernel<<<256, 256, 0, stream>>>(agg, N_NODES * DHID);
        edge_msg_kernel<64><<<1024, 256, 0, stream>>>(
            hc, eattr, srcI, dstI,
            WeL + (size_t)i * EDGE_DIM * DHID, beL + i * DHID,
            agg, edgeTiles);
        node_mlp_kernel<64><<<mlpBlocks, 256, 0, stream>>>(
            hc, agg, epsL + i,
            WLa + (size_t)i * DHID * DHID, bLa + i * DHID,
            WLb + (size_t)i * DHID * DHID, bLb + i * DHID,
            hn, nodeTiles);
        float* tmp = hc; hc = hn; hn = tmp;
    }

    // ---- pooling + head ----
    fill_zero_kernel<<<64, 256, 0, stream>>>((float*)gmaxU,
                                             N_GRAPHS * DHID * 2 + N_GRAPHS);
    pool_kernel<<<2048, 256, 0, stream>>>(hc, batch, gmaxU, gsum, counts);
    pool_finalize_kernel<<<(N_GRAPHS * DHID + 255) / 256, 256, 0, stream>>>(
        gmaxU, gsum, counts, gmat);
    head_kernel<<<N_GRAPHS, 256, 0, stream>>>(gmat, Wl, bl, Wo, bo, out);
}
